// DifferentiableTreeDense_23098334118430
// MI455X (gfx1250) — matrix operand order repaired and checked by reading, not compiled, not run on hardware
//
#include <hip/hip_runtime.h>

typedef unsigned short ushort_t;
typedef __attribute__((ext_vector_type(16))) __bf16 bf16x16;
typedef __attribute__((ext_vector_type(8)))  float  f32x8;

union Frag { uint4 q[2]; bf16x16 v; };

__device__ __forceinline__ ushort_t f2bf(float f) {
    unsigned int u = __float_as_uint(f);
    u = u + 0x7FFFu + ((u >> 16) & 1u);   // round-to-nearest-even
    return (ushort_t)(u >> 16);
}

// async global->LDS 16B copy (CDNA5, ASYNCcnt-tracked). ldsOff = AS(3) byte offset.
__device__ __forceinline__ void async_ld16(unsigned ldsOff, const ushort_t* g) {
    asm volatile("global_load_async_to_lds_b128 %0, %1, off"
                 :: "v"(ldsOff), "v"(g) : "memory");
}

// ---------------- prepack kernels ----------------

__global__ __launch_bounds__(256) void cvt_x_kernel(const float* __restrict__ x,
                                                    ushort_t* __restrict__ xb) {
    int i = blockIdx.x * blockDim.x + threadIdx.x;      // i indexes float4
    float4 v = reinterpret_cast<const float4*>(x)[i];
    ushort_t r[4] = { f2bf(v.x), f2bf(v.y), f2bf(v.z), f2bf(v.w) };
    *reinterpret_cast<uint2*>(xb + i * 4) = *reinterpret_cast<uint2*>(r);
}

// leaf_weights [64][1024][128] f32 -> Wb [8192 cols][1024 k] bf16 (transpose per leaf)
__global__ __launch_bounds__(256) void pack_w_kernel(const float* __restrict__ W,
                                                     ushort_t* __restrict__ Wb) {
    __shared__ float tile[32][33];
    const int l  = blockIdx.z;
    const int o0 = blockIdx.x * 32;
    const int k0 = blockIdx.y * 32;
    const int tx = threadIdx.x, ty = threadIdx.y;   // 32 x 8
    const float* Wl = W + (size_t)l * 1024 * 128;
#pragma unroll
    for (int j = 0; j < 4; ++j) {
        int k = k0 + ty + j * 8;
        tile[ty + j * 8][tx] = Wl[k * 128 + o0 + tx];
    }
    __syncthreads();
    ushort_t* outp = Wb + (size_t)(l * 128) * 1024;
#pragma unroll
    for (int j = 0; j < 4; ++j) {
        int o = o0 + ty + j * 8;
        outp[(size_t)o * 1024 + k0 + tx] = f2bf(tile[tx][ty + j * 8]);
    }
}

// routing_weights [63][1024][2] f32 -> Rb [128][1024] bf16, cols 126/127 zero
__global__ __launch_bounds__(256) void pack_r_kernel(const float* __restrict__ R,
                                                     ushort_t* __restrict__ Rb) {
    int idx = blockIdx.x * blockDim.x + threadIdx.x;    // n*1024 + k
    int n = idx >> 10, k = idx & 1023;
    float v = (n < 126) ? R[(size_t)(n >> 1) * 2048 + k * 2 + (n & 1)] : 0.0f;
    Rb[idx] = f2bf(v);
}

// ---------------- softmax + tree path product ----------------
__global__ __launch_bounds__(256) void tree_probs_kernel(const float* __restrict__ logits,
                                                         float* __restrict__ lp) {
    int idx = blockIdx.x * blockDim.x + threadIdx.x;    // b*64 + l
    int b = idx >> 6, l = idx & 63;
    const float* lg = logits + (size_t)b * 128;
    float p = 1.0f;
    int node = 0;
#pragma unroll
    for (int d = 0; d < 6; ++d) {
        int bit = (l >> (5 - d)) & 1;
        float z0 = lg[node * 2], z1 = lg[node * 2 + 1];
        float zd = bit ? z1 : z0;
        float zo = bit ? z0 : z1;
        p *= 1.0f / (1.0f + __expf(zo - zd));
        node = 2 * node + 1 + bit;
    }
    lp[idx] = p;
}

// ---------------- WMMA GEMM (bf16 in, f32 accum), async double-buffered ----------------
// A [M][K] bf16 row-major, B [Ncols][K] bf16 row-major (pre-transposed).
// Block tile 256x128 (8 waves, wave tile 64x64 = 4x4 of 16x16), K staged 32 wide,
// double-buffered LDS fed by global_load_async_to_lds_b128.
template <bool TREE>
__global__ __launch_bounds__(256) void gemm_bf16_kernel(
    const ushort_t* __restrict__ A, const ushort_t* __restrict__ B,
    const float* __restrict__ probs, const float* __restrict__ bias,
    float* __restrict__ out, int ldOut, int K)
{
    __shared__ ushort_t Asd[2][256 * 32];   // 2 x 16 KB
    __shared__ ushort_t Bsd[2][128 * 32];   // 2 x  8 KB
    __shared__ float probS[256];
    __shared__ float biasS[128];

    const int t    = threadIdx.x;
    const int lane = t & 31;
    const int w    = t >> 5;
    const int blockM = blockIdx.y * 256;
    const int blockN = blockIdx.x * 128;
    const int waveM  = (w >> 1) * 64;       // 4 wave-rows
    const int waveN  = (w & 1) * 64;        // 2 wave-cols

    if (TREE) {
        probS[t] = probs[(size_t)(blockM + t) * 64 + blockIdx.x]; // one leaf per N-block
        if (t < 128) biasS[t] = bias[blockIdx.x * 128 + t];
    }

    f32x8 acc[4][4];
#pragma unroll
    for (int mt = 0; mt < 4; ++mt)
#pragma unroll
        for (int nt = 0; nt < 4; ++nt)
#pragma unroll
            for (int r = 0; r < 8; ++r) acc[mt][nt][r] = 0.0f;

    // Per-thread async copy plan (per K-step): A row t -> 4x16B, B row t>>1 half -> 2x16B.
    const ushort_t* Aga = A + (size_t)(blockM + t) * K;
    const ushort_t* Bga = B + (size_t)(blockN + (t >> 1)) * K + (t & 1) * 16;
    const unsigned AsOff = (unsigned)(unsigned long long)(&Asd[0][0]) + (unsigned)t * 64u;
    const unsigned BsOff = (unsigned)(unsigned long long)(&Bsd[0][0]) +
                           (unsigned)(t >> 1) * 64u + (unsigned)(t & 1) * 32u;
    const unsigned ABUF = 256 * 32 * 2;     // bytes per A buffer
    const unsigned BBUF = 128 * 32 * 2;     // bytes per B buffer

    const int steps = K >> 5;
    // prologue: fill buffer 0
    {
#pragma unroll
        for (int j = 0; j < 4; ++j) async_ld16(AsOff + j * 16u, Aga + j * 8);
#pragma unroll
        for (int j = 0; j < 2; ++j) async_ld16(BsOff + j * 16u, Bga + j * 8);
    }

    const int half = lane >> 4;
    const int mr   = lane & 15;

    for (int i = 0; i < steps; ++i) {
        const int buf = i & 1;
        __syncthreads();   // all waves done reading buf(i-1) -> safe to overwrite
        if (i + 1 < steps) {
            const int kk = (i + 1) << 5;
            const unsigned nb = (unsigned)((i + 1) & 1);
#pragma unroll
            for (int j = 0; j < 4; ++j)
                async_ld16(AsOff + nb * ABUF + j * 16u, Aga + kk + j * 8);
#pragma unroll
            for (int j = 0; j < 2; ++j)
                async_ld16(BsOff + nb * BBUF + j * 16u, Bga + kk + j * 8);
            asm volatile("s_wait_asynccnt 0x6" ::: "memory");  // group i retired
        } else {
            asm volatile("s_wait_asynccnt 0x0" ::: "memory");
        }
        __syncthreads();   // buf(i) visible to all waves

        const ushort_t* Ab = Asd[buf];
        const ushort_t* Bb = Bsd[buf];

        // A frag: elems 0-7 -> K=half*8+0..7 ; elems 8-15 -> K=16+half*8+0..7
        Frag af[4];
#pragma unroll
        for (int mt = 0; mt < 4; ++mt) {
            const ushort_t* p = Ab + (waveM + mt * 16 + mr) * 32;
            af[mt].q[0] = *reinterpret_cast<const uint4*>(p + half * 8);
            af[mt].q[1] = *reinterpret_cast<const uint4*>(p + 16 + half * 8);
        }
        // B frag: elems 0-7 -> K=half*8+0..7 ; elems 8-15 -> K=16+half*8+0..7
        Frag bfr[4];
#pragma unroll
        for (int nt = 0; nt < 4; ++nt) {
            const ushort_t* p = Bb + (waveN + nt * 16 + mr) * 32;
            bfr[nt].q[0] = *reinterpret_cast<const uint4*>(p + half * 8);
            bfr[nt].q[1] = *reinterpret_cast<const uint4*>(p + 16 + half * 8);
        }
#pragma unroll
        for (int mt = 0; mt < 4; ++mt)
#pragma unroll
            for (int nt = 0; nt < 4; ++nt)
                acc[mt][nt] = __builtin_amdgcn_wmma_f32_16x16x32_bf16(
                    false, af[mt].v, false, bfr[nt].v,
                    (short)0, acc[mt][nt], false, false);
    }

    // epilogue. C/D layout: M = half*8 + r, N = lane&15
#pragma unroll
    for (int mt = 0; mt < 4; ++mt) {
#pragma unroll
        for (int nt = 0; nt < 4; ++nt) {
            const int localN = waveN + nt * 16 + mr;
            const int gN = blockN + localN;
#pragma unroll
            for (int r = 0; r < 8; ++r) {
                const int localM = waveM + mt * 16 + half * 8 + r;
                const int gM = blockM + localM;
                float v = acc[mt][nt][r];
                if (TREE) v = probS[localM] * (v + biasS[localN]);
                out[(size_t)gM * ldOut + gN] = v;
            }
        }
    }
}

// ---------------- launch ----------------
extern "C" void kernel_launch(void* const* d_in, const int* in_sizes, int n_in,
                              void* d_out, int out_size, void* d_ws, size_t ws_size,
                              hipStream_t stream) {
    (void)in_sizes; (void)n_in; (void)out_size; (void)ws_size;
    const float* x  = (const float*)d_in[0];   // [8192][1024]
    const float* rw = (const float*)d_in[1];   // [63][1024][2]
    const float* lw = (const float*)d_in[2];   // [64][1024][128]
    const float* lb = (const float*)d_in[3];   // [64][128]
    float* out = (float*)d_out;                // [8192][8192]

    char* ws = (char*)d_ws;
    ushort_t* xb     = (ushort_t*)(ws);                 // 16,777,216 B
    ushort_t* Wb     = (ushort_t*)(ws + 16777216);      // 16,777,216 B
    ushort_t* Rb     = (ushort_t*)(ws + 33554432);      //    262,144 B
    float*    logits = (float*)   (ws + 33816576);      //  4,194,304 B
    float*    lp     = (float*)   (ws + 38010880);      //  2,097,152 B

    cvt_x_kernel<<<8192, 256, 0, stream>>>(x, xb);
    pack_w_kernel<<<dim3(4, 32, 64), dim3(32, 8), 0, stream>>>(lw, Wb);
    pack_r_kernel<<<512, 256, 0, stream>>>(rw, Rb);

    // routing logits: [8192x1024] @ [1024x128]
    gemm_bf16_kernel<false><<<dim3(1, 32), 256, 0, stream>>>(
        xb, Rb, nullptr, nullptr, logits, 128, 1024);

    tree_probs_kernel<<<2048, 256, 0, stream>>>(logits, lp);

    // main GEMM [8192x1024] @ [1024x8192], fused prob-scale + bias
    gemm_bf16_kernel<true><<<dim3(64, 32), 256, 0, stream>>>(
        xb, Wb, lp, lb, out, 8192, 1024);
}
